// PartitionAttentionLayer_4286377361635
// MI455X (gfx1250) — matrix operand order repaired and checked by reading, not compiled, not run on hardware
//
#include <hip/hip_runtime.h>
#include <hip/hip_bf16.h>
#include <math.h>

typedef __attribute__((ext_vector_type(16))) _Float16 v16h;
typedef __attribute__((ext_vector_type(8)))  float    v8f;
typedef __attribute__((ext_vector_type(2)))  _Float16 h2v;
typedef unsigned int uint;
typedef __attribute__((ext_vector_type(4))) uint u32x4;
typedef __attribute__((ext_vector_type(4))) int  i32x4v;
typedef __attribute__((ext_vector_type(8))) int  i32x8v;

#if defined(__has_builtin)
#if __has_builtin(__builtin_amdgcn_tensor_load_to_lds) && __has_builtin(__builtin_amdgcn_s_wait_tensorcnt)
#define HAVE_TDM 1
#endif
#endif

// ---------------- constants ----------------
#define NB      32
#define CCH     256
#define HH      64
#define WW      64
#define NHEAD   8
#define HD      32

// LDS strides (elements), padded to dodge bank conflicts, f16 ones even (4B pair align)
#define XCOL_S  65    // f32, channel-major x staging: [256 c][65] (TDM pad: 1 dword per 64)
#define XLN_S   264   // f16 (also h2)
#define QKV_S   776   // f16
#define P_S     72    // f16
#define O_S     264   // f16
#define PJ_S    260   // f32
#define M1_S    1032  // f16

// LDS regions (bytes)
#define REGA_OFF   0                         // xln(33792) / pbuf(73728) / h2(33792)
#define OBUF_OFF   73728                     // 33792
#define REGB_OFF   107520                    // xcol(66560)/qkv(99328)/proj(66560)/mlp1(132096)
#define SMEM_BYTES (107520 + 132096)         // 239616 <= 320KB/WGP

__device__ __forceinline__ v8f wmma16(v16h a, v16h b, v8f c) {
  return __builtin_amdgcn_wmma_f32_16x16x32_f16(false, a, false, b, (short)0, c, false, false);
}

__device__ __forceinline__ v8f zero8() {
  v8f z;
#pragma unroll
  for (int i = 0; i < 8; ++i) z[i] = 0.0f;
  return z;
}

// A tile 16x32 f16 (M x K), source row-major: base=elem(m=0,k=0), row stride ldm, k contiguous.
// Contiguous k order used here (not the ISA table order): lane m=lane&15, half=lane>>4; VGPR j(0..7): K=half*16+2j(,+1)
__device__ __forceinline__ v16h load_A(const _Float16* base, int ldm) {
  int lane = threadIdx.x & 31;
  int m = lane & 15, hf = lane >> 4;
  const _Float16* row = base + m * ldm;
  v16h a;
#pragma unroll
  for (int j = 0; j < 4; ++j) {
    h2v p0 = *(const h2v*)(row + hf * 16 + 2 * j);
    h2v p1 = *(const h2v*)(row + hf * 16 + 8 + 2 * j);
    a[2 * j] = p0.x; a[2 * j + 1] = p0.y;
    a[8 + 2 * j] = p1.x; a[8 + 2 * j + 1] = p1.y;
  }
  return a;
}

// B tile 32x16 (K x N) where B(k,n) = base[n*ldn + k] (weights stored [N][K] row-major).
// Contiguous k order used here (not the ISA table order): lane n=lane&15, half=lane>>4; VGPR j: K = half*16 + 2j (,+1)
__device__ __forceinline__ v16h load_Bk(const _Float16* base, int ldn) {
  int lane = threadIdx.x & 31;
  int n = lane & 15, hf = lane >> 4;
  const _Float16* row = base + n * ldn + hf * 16;
  v16h b;
#pragma unroll
  for (int j = 0; j < 8; ++j) {
    h2v p = *(const h2v*)(row + 2 * j);
    b[2 * j] = p.x; b[2 * j + 1] = p.y;
  }
  return b;
}

// Generic B tile 32x16: B(k,n) = base[k*strideK + n*strideN]
__device__ __forceinline__ v16h load_Bg(const _Float16* base, int strideK, int strideN) {
  int lane = threadIdx.x & 31;
  int n = lane & 15, hf = lane >> 4;
  const _Float16* col = base + n * strideN + (hf * 16) * strideK;
  v16h b;
#pragma unroll
  for (int j = 0; j < 8; ++j) {
    b[2 * j]     = col[(2 * j) * strideK];
    b[2 * j + 1] = col[(2 * j + 1) * strideK];
  }
  return b;
}

// LayerNorm over 64 rows x 256 cols; 4 threads per row (xor-shuffle 1,2 within wave).
// Element (t,c) at src[t*ts + c*cs].
__device__ __forceinline__ void ln_rows(const float* src, int ts, int cs,
                                        _Float16* dst, int ds,
                                        const float* gg, const float* bb) {
  int t = threadIdx.x >> 2, q = threadIdx.x & 3;
  const float* base = src + t * ts;
  float s = 0.f;
  for (int i = 0; i < 64; ++i) s += base[(q * 64 + i) * cs];
  s += __shfl_xor(s, 1);
  s += __shfl_xor(s, 2);
  float mean = s * (1.0f / 256.0f);
  float v = 0.f;
  for (int i = 0; i < 64; ++i) { float d = base[(q * 64 + i) * cs] - mean; v += d * d; }
  v += __shfl_xor(v, 1);
  v += __shfl_xor(v, 2);
  float rinv = rsqrtf(v * (1.0f / 256.0f) + 1e-5f);
  for (int i = 0; i < 64; ++i) {
    int c = q * 64 + i;
    dst[t * ds + c] = (_Float16)((base[c * cs] - mean) * rinv * gg[c] + bb[c]);
  }
}

// fp32 -> f16 weight conversion
__global__ void cvt_f16_kernel(const float* __restrict__ s, _Float16* __restrict__ d, int n) {
  int i = blockIdx.x * 256 + threadIdx.x;
  if (i < n) d[i] = (_Float16)s[i];
}

__launch_bounds__(256, 1)
__global__ void swin_block_kernel(
    const float* __restrict__ x,
    const float* __restrict__ ln1_g, const float* __restrict__ ln1_b,
    const float* __restrict__ qkv_b, const float* __restrict__ proj_b,
    const float* __restrict__ rel_bias, const int* __restrict__ rel_index,
    const float* __restrict__ ln2_g, const float* __restrict__ ln2_b,
    const float* __restrict__ b1, const float* __restrict__ b2,
    const _Float16* __restrict__ qkv_wh, const _Float16* __restrict__ proj_wh,
    const _Float16* __restrict__ w1h, const _Float16* __restrict__ w2h,
    float* __restrict__ out) {
  extern __shared__ char smem[];
  _Float16* xln    = (_Float16*)(smem + REGA_OFF);   // LN1 out  [64][XLN_S]
  _Float16* pbuf   = (_Float16*)(smem + REGA_OFF);   // P probs  8 x [64][P_S]   (after QKV barrier)
  _Float16* h2buf  = (_Float16*)(smem + REGA_OFF);   // LN2 out  [64][XLN_S]     (after attn barrier)
  _Float16* obuf   = (_Float16*)(smem + OBUF_OFF);   // attn out [64][O_S]
  float*    xcol   = (float*)   (smem + REGB_OFF);   // raw x    [256 c][XCOL_S] (channel-major)
  _Float16* qkvbuf = (_Float16*)(smem + REGB_OFF);   // qkv      [64][QKV_S]
  float*    projbuf= (float*)   (smem + REGB_OFF);   // proj out [64][PJ_S]
  _Float16* m1buf  = (_Float16*)(smem + REGB_OFF);   // mlp hid  [64][M1_S]

  const int tid  = threadIdx.x;
  const int wv   = tid >> 5;      // wave id == head id
  const int lane = tid & 31;
  const int ln15 = lane & 15;
  const int hf   = lane >> 4;

  const int wid  = blockIdx.x;
  const int bidx = wid >> 6;
  const int g    = wid & 63;
  const int gy   = g >> 3;
  const int gx   = g & 7;

  // warm the caches for the f16 weight panels (global_prefetch_b8)
  __builtin_prefetch(qkv_wh + (size_t)tid * 768, 0, 1);
  __builtin_prefetch(w1h + (size_t)tid * 1024, 0, 1);

  // ---------- Phase 0: window [8px][8py][256c] -> LDS xcol[c][65] ----------
  const float* xb = x + (size_t)bidx * CCH * HH * WW + (size_t)(gy * 8) * WW + gx * 8;
#ifdef HAVE_TDM
  if (tid < 32) {  // wave 0 drives the Tensor Data Mover
    unsigned long long gaddr = (unsigned long long)(const void*)xb;
    u32x4 g0;
    g0[0] = 1u;                                        // count=1 valid descriptor
    g0[1] = (uint)REGB_OFF;                            // lds_addr (bytes)
    g0[2] = (uint)(gaddr & 0xFFFFFFFFull);             // global_addr lo
    g0[3] = (uint)((gaddr >> 32) & 0x1FFFFFFull) | (2u << 30);  // global_addr hi, type=2
    i32x8v g1;
    // data_size=4B(2), pad_enable, pad_interval=64dw(code 5), pad_amount=1dw(code 0)
    g1[0] = (int)((2u << 16) | (1u << 20) | (5u << 22));
    g1[1] = (int)(64u << 16);                          // tensor_dim0[15:0]=64 (row length)
    g1[2] = (int)(64u << 16);                          // tensor_dim0 hi=0; tensor_dim1[15:0]=64
    g1[3] = (int)(8u << 16);                           // tensor_dim1 hi=0; tile_dim0=8 (px)
    g1[4] = (int)(8u | (256u << 16));                  // tile_dim1=8 (py), tile_dim2=256 (c)
    g1[5] = 64;                                        // tensor_dim0_stride lo32 = 64 (row stride)
    g1[6] = (int)(4096u << 16);                        // t0s hi=0; tensor_dim1_stride[15:0]=4096 (chan stride)
    g1[7] = 0;                                         // tensor_dim1_stride hi = 0
    i32x4v g2;
    g2[0] = 256;                                       // tensor_dim2 = 256
    g2[1] = 1;                                         // tensor_dim3 (unused)
    g2[2] = 4096;                                      // tensor_dim2_stride lo (unused for 3D)
    g2[3] = 0;                                         // stride hi; tile_dim3=0
    i32x4v g3; g3[0] = 0; g3[1] = 0; g3[2] = 0; g3[3] = 0;
#if __clang_major__ >= 23
    i32x8v gz;
#pragma unroll
    for (int i = 0; i < 8; ++i) gz[i] = 0;
    __builtin_amdgcn_tensor_load_to_lds(g0, g1, g2, g3, gz, 0);
#else
    __builtin_amdgcn_tensor_load_to_lds(g0, g1, g2, g3, 0);
#endif
    __builtin_amdgcn_s_wait_tensorcnt(0);
  }
#else
  for (int idx = tid; idx < 64 * 256; idx += 256) {
    int c = idx >> 6, t = idx & 63;
    int py = t >> 3, px = t & 7;
    xcol[c * XCOL_S + t] = xb[(size_t)c * (HH * WW) + py * WW + px];
  }
#endif
  __syncthreads();

  // ---------- Phase 1: LN1 -> xln f16 (xcol is channel-major: (t,c) at c*65+t) ----------
  ln_rows(xcol, 1, XCOL_S, xln, XLN_S, ln1_g, ln1_b);
  __syncthreads();

  // ---------- Phase 2: QKV GEMM; wave w computes q/k/v cols for head w ----------
  for (int part = 0; part < 3; ++part) {
    for (int ntl = 0; ntl < 2; ++ntl) {
      int ncol = part * 256 + wv * 32 + ntl * 16;
      v8f acc[4];
#pragma unroll
      for (int mt = 0; mt < 4; ++mt) acc[mt] = zero8();
      for (int kk = 0; kk < 8; ++kk) {
        v16h bm = load_Bk(qkv_wh + (size_t)ncol * 256 + kk * 32, 256);
#pragma unroll
        for (int mt = 0; mt < 4; ++mt) {
          v16h am = load_A(xln + (mt * 16) * XLN_S + kk * 32, XLN_S);
          acc[mt] = wmma16(am, bm, acc[mt]);
        }
      }
      float bias = qkv_b[ncol + ln15];
#pragma unroll
      for (int mt = 0; mt < 4; ++mt)
#pragma unroll
        for (int r = 0; r < 8; ++r)
          qkvbuf[(mt * 16 + r + 8 * hf) * QKV_S + ncol + ln15] = (_Float16)(acc[mt][r] + bias);
    }
  }
  __syncthreads();  // needed: pbuf reuses xln region

  // ---------- Phase 3: attention, one head per wave ----------
  {
    const _Float16* qbase = qkvbuf + wv * 32;
    const _Float16* kbase = qkvbuf + 256 + wv * 32;
    const _Float16* vbase = qkvbuf + 512 + wv * 32;
    _Float16* pw = pbuf + wv * (64 * P_S);

    v8f att[4][4];
#pragma unroll
    for (int mt = 0; mt < 4; ++mt) {
      v16h qa = load_A(qbase + (mt * 16) * QKV_S, QKV_S);
#pragma unroll
      for (int nt = 0; nt < 4; ++nt) {
        v16h kb = load_Bk(kbase + (nt * 16) * QKV_S, QKV_S);
        att[mt][nt] = wmma16(qa, kb, zero8());
      }
    }
    const float scale = 0.17677669529663689f;  // 1/sqrt(32)
#pragma unroll
    for (int mt = 0; mt < 4; ++mt) {
#pragma unroll
      for (int r = 0; r < 8; ++r) {
        int t = mt * 16 + r + 8 * hf;
        float mx = -1e30f;
#pragma unroll
        for (int nt = 0; nt < 4; ++nt) {
          int s = nt * 16 + ln15;
          float bv = rel_bias[rel_index[t * 64 + s] * NHEAD + wv];
          float v = att[mt][nt][r] * scale + bv;
          att[mt][nt][r] = v;
          mx = fmaxf(mx, v);
        }
        mx = fmaxf(mx, __shfl_xor(mx, 1));
        mx = fmaxf(mx, __shfl_xor(mx, 2));
        mx = fmaxf(mx, __shfl_xor(mx, 4));
        mx = fmaxf(mx, __shfl_xor(mx, 8));
        float sum = 0.f;
#pragma unroll
        for (int nt = 0; nt < 4; ++nt) {
          float e = __expf(att[mt][nt][r] - mx);
          att[mt][nt][r] = e;
          sum += e;
        }
        sum += __shfl_xor(sum, 1);
        sum += __shfl_xor(sum, 2);
        sum += __shfl_xor(sum, 4);
        sum += __shfl_xor(sum, 8);
        float inv = 1.0f / sum;
#pragma unroll
        for (int nt = 0; nt < 4; ++nt)
          pw[t * P_S + nt * 16 + ln15] = (_Float16)(att[mt][nt][r] * inv);
      }
    }
    // P @ V  (K = 64 over s, two K-steps)
#pragma unroll
    for (int nt2 = 0; nt2 < 2; ++nt2) {
      v16h vb0 = load_Bg(vbase + nt2 * 16, QKV_S, 1);
      v16h vb1 = load_Bg(vbase + 32 * QKV_S + nt2 * 16, QKV_S, 1);
#pragma unroll
      for (int mt = 0; mt < 4; ++mt) {
        v16h p0 = load_A(pw + (mt * 16) * P_S, P_S);
        v16h p1 = load_A(pw + (mt * 16) * P_S + 32, P_S);
        v8f o = wmma16(p0, vb0, zero8());
        o = wmma16(p1, vb1, o);
#pragma unroll
        for (int r = 0; r < 8; ++r)
          obuf[(mt * 16 + r + 8 * hf) * O_S + wv * 32 + nt2 * 16 + ln15] = (_Float16)o[r];
      }
    }
  }
  __syncthreads();  // obuf complete; qkvbuf region reused as projbuf

  // ---------- Phase 4: output projection ----------
  for (int ntl = 0; ntl < 2; ++ntl) {
    int ncol = (wv * 2 + ntl) * 16;
    v8f acc[4];
#pragma unroll
    for (int mt = 0; mt < 4; ++mt) acc[mt] = zero8();
    for (int kk = 0; kk < 8; ++kk) {
      v16h bm = load_Bk(proj_wh + (size_t)ncol * 256 + kk * 32, 256);
#pragma unroll
      for (int mt = 0; mt < 4; ++mt) {
        v16h am = load_A(obuf + (mt * 16) * O_S + kk * 32, O_S);
        acc[mt] = wmma16(am, bm, acc[mt]);
      }
    }
    float bias = proj_b[ncol + ln15];
#pragma unroll
    for (int mt = 0; mt < 4; ++mt)
#pragma unroll
      for (int r = 0; r < 8; ++r)
        projbuf[(mt * 16 + r + 8 * hf) * PJ_S + ncol + ln15] = acc[mt][r] + bias;
  }
  __syncthreads();

  // ---------- Phase 5: LN2 -> h2 f16 (reuses region A) ----------
  ln_rows(projbuf, PJ_S, 1, h2buf, XLN_S, ln2_g, ln2_b);
  __syncthreads();

  // ---------- Phase 6: MLP1 + exact GELU -> m1buf f16 ----------
  for (int ntl = 0; ntl < 8; ++ntl) {
    int ncol = (wv * 8 + ntl) * 16;
    v8f acc[4];
#pragma unroll
    for (int mt = 0; mt < 4; ++mt) acc[mt] = zero8();
    for (int kk = 0; kk < 8; ++kk) {
      v16h bm = load_Bk(w1h + (size_t)ncol * 256 + kk * 32, 256);
#pragma unroll
      for (int mt = 0; mt < 4; ++mt) {
        v16h am = load_A(h2buf + (mt * 16) * XLN_S + kk * 32, XLN_S);
        acc[mt] = wmma16(am, bm, acc[mt]);
      }
    }
    float bias = b1[ncol + ln15];
#pragma unroll
    for (int mt = 0; mt < 4; ++mt)
#pragma unroll
      for (int r = 0; r < 8; ++r) {
        float vv = acc[mt][r] + bias;
        float gl = 0.5f * vv * (1.0f + erff(vv * 0.70710678118654752f));
        m1buf[(mt * 16 + r + 8 * hf) * M1_S + ncol + ln15] = (_Float16)gl;
      }
  }
  __syncthreads();

  // ---------- Phase 7: MLP2 + departition store (contiguous 8 floats per lane-row) ----------
  for (int ntl = 0; ntl < 2; ++ntl) {
    int ncol = (wv * 2 + ntl) * 16;
    v8f acc[4];
#pragma unroll
    for (int mt = 0; mt < 4; ++mt) acc[mt] = zero8();
    for (int kk = 0; kk < 32; ++kk) {
      v16h bm = load_Bk(w2h + (size_t)ncol * 1024 + kk * 32, 1024);
#pragma unroll
      for (int mt = 0; mt < 4; ++mt) {
        v16h am = load_A(m1buf + (mt * 16) * M1_S + kk * 32, M1_S);
        acc[mt] = wmma16(am, bm, acc[mt]);
      }
    }
    float bias = b2[ncol + ln15];
    int cch = ncol + ln15;
#pragma unroll
    for (int mt = 0; mt < 4; ++mt) {
      int py = mt * 2 + hf;   // row t = mt*16 + 8*hf + r  ->  py = t>>3, px = r
      float* op = out + (((size_t)bidx * CCH + cch) * HH + gy * 8 + py) * WW + gx * 8;
      float4 lo = make_float4(acc[mt][0] + bias, acc[mt][1] + bias, acc[mt][2] + bias, acc[mt][3] + bias);
      float4 hi = make_float4(acc[mt][4] + bias, acc[mt][5] + bias, acc[mt][6] + bias, acc[mt][7] + bias);
      *(float4*)(op) = lo;
      *(float4*)(op + 4) = hi;
    }
  }
}

extern "C" void kernel_launch(void* const* d_in, const int* in_sizes, int n_in,
                              void* d_out, int out_size, void* d_ws, size_t ws_size,
                              hipStream_t stream) {
  const float* x        = (const float*)d_in[0];
  const float* ln1_g    = (const float*)d_in[1];
  const float* ln1_b    = (const float*)d_in[2];
  const float* qkv_w    = (const float*)d_in[3];
  const float* qkv_b    = (const float*)d_in[4];
  const float* proj_w   = (const float*)d_in[5];
  const float* proj_b   = (const float*)d_in[6];
  const float* rel_bias = (const float*)d_in[7];
  const float* ln2_g    = (const float*)d_in[8];
  const float* ln2_b    = (const float*)d_in[9];
  const float* w1       = (const float*)d_in[10];
  const float* b1       = (const float*)d_in[11];
  const float* w2       = (const float*)d_in[12];
  const float* b2       = (const float*)d_in[13];
  const int*   rel_index= (const int*)d_in[14];
  float* out = (float*)d_out;

  _Float16* wsp = (_Float16*)d_ws;
  _Float16* qkv_wh  = wsp;                // 768*256   = 196608
  _Float16* proj_wh = wsp + 196608;       // 256*256   =  65536
  _Float16* w1h     = wsp + 262144;       // 1024*256  = 262144
  _Float16* w2h     = wsp + 524288;       // 256*1024  = 262144

  cvt_f16_kernel<<<(196608 + 255) / 256, 256, 0, stream>>>(qkv_w, qkv_wh, 196608);
  cvt_f16_kernel<<<(65536 + 255) / 256, 256, 0, stream>>>(proj_w, proj_wh, 65536);
  cvt_f16_kernel<<<(262144 + 255) / 256, 256, 0, stream>>>(w1, w1h, 262144);
  cvt_f16_kernel<<<(262144 + 255) / 256, 256, 0, stream>>>(w2, w2h, 262144);

  (void)hipFuncSetAttribute((const void*)swin_block_kernel,
                            hipFuncAttributeMaxDynamicSharedMemorySize, SMEM_BYTES);
  swin_block_kernel<<<NB * 64, 256, SMEM_BYTES, stream>>>(
      x, ln1_g, ln1_b, qkv_b, proj_b, rel_bias, rel_index, ln2_g, ln2_b,
      b1, b2, qkv_wh, proj_wh, w1h, w2h, out);
}
